// RecurrentPreExpertRouter_39410619908671
// MI455X (gfx1250) — matrix operand order repaired and checked by reading, not compiled, not run on hardware
//
#include <hip/hip_runtime.h>
#include <cstddef>

// ---------------- problem constants ----------------
#define BATCH 4
#define SEQ   4096
#define HID   2048
#define NEXP  64
#define DDIM  256
#define NTOK  (BATCH*SEQ)          // 16384
#define KSEL  4096                 // ceil(0.25 * NTOK)

// B slab staging in LDS
#define BROWS 320                  // 256 ts rows + 64 router rows
#define BPAD  80                   // bytes per 32-k bf16 row (64B data + 16B pad; bank-conflict free)

// Toggle the CDNA5 async global->LDS DMA path (ASYNCcnt). Fallback = VGPR copies.
#define USE_ASYNC_LDS 1

typedef __attribute__((ext_vector_type(16))) __bf16 v16bf;
typedef __attribute__((ext_vector_type(8)))  __bf16 v8bf;
typedef __attribute__((ext_vector_type(8)))  float  v8f;

// flags for apply/softmax kernel
#define FL_MASK  1
#define FL_WSUM  2
#define FL_WRITE 4

// ---------------- tiny utility kernels ----------------
__global__ void zero_f32_kernel(float* p, int n) {
    int i = blockIdx.x * 256 + threadIdx.x;
    if (i < n) p[i] = 0.0f;
}

__global__ void cvt_bf16_kernel(const float* __restrict__ src, __bf16* __restrict__ dst, int n) {
    int i = blockIdx.x * 256 + threadIdx.x;
    if (i < n) dst[i] = (__bf16)src[i];
}

// ---------------- async copy helpers ----------------
__device__ __forceinline__ void async_copy16(const void* g, void* l) {
#if USE_ASYNC_LDS
    // per-lane 16B global->LDS DMA; tracked by ASYNCcnt (ISA 08_async_tensor §4)
    unsigned           lds = (unsigned)(unsigned long long)(size_t)l;   // low 32b = LDS addr
    unsigned long long ga  = (unsigned long long)(size_t)g;
    asm volatile("global_load_async_to_lds_b128 %0, %1, off"
                 :: "v"(lds), "v"(ga) : "memory");
#else
    *(float4*)l = *(const float4*)g;
#endif
}

__device__ __forceinline__ void async_wait_keep5() {
#if USE_ASYNC_LDS
    asm volatile("s_wait_asynccnt 0x5" ::: "memory");   // 5 newer chunks may stay in flight
#endif
}
__device__ __forceinline__ void async_wait_all() {
#if USE_ASYNC_LDS
    asm volatile("s_wait_asynccnt 0x0" ::: "memory");
#endif
}

// cooperative issue of one 320x32 bf16 B-slab into an LDS buffer (1280 x 16B chunks)
__device__ __forceinline__ void load_b_tile(const __bf16* __restrict__ Wall, int k0,
                                            unsigned char* buf, int tid) {
#pragma unroll
    for (int t = 0; t < 5; ++t) {
        int c    = tid + t * 256;        // 0..1279
        int row  = c >> 2;
        int part = c & 3;
        const void* g = (const char*)(Wall + (size_t)row * HID + k0) + part * 16;
        void*       l = buf + row * BPAD + part * 16;
        async_copy16(g, l);
    }
}

// ---------------- register-resident fragment helpers ----------------
struct AF { float4 f0, f1, f2, f3; };

__device__ __forceinline__ AF load_a(const float* __restrict__ hrow, int k0, int half) {
    AF r;
    const float4* hp  = (const float4*)(hrow + k0 + half * 8);
    const float4* hp2 = (const float4*)(hrow + k0 + half * 8 + 16);
    r.f0 = hp[0]; r.f1 = hp[1]; r.f2 = hp2[0]; r.f3 = hp2[1];
    return r;
}

__device__ __forceinline__ v16bf cvt_a(const AF& r) {
    v16bf a;
    a[0]=(__bf16)r.f0.x; a[1]=(__bf16)r.f0.y; a[2]=(__bf16)r.f0.z; a[3]=(__bf16)r.f0.w;
    a[4]=(__bf16)r.f1.x; a[5]=(__bf16)r.f1.y; a[6]=(__bf16)r.f1.z; a[7]=(__bf16)r.f1.w;
    a[8]=(__bf16)r.f2.x; a[9]=(__bf16)r.f2.y; a[10]=(__bf16)r.f2.z; a[11]=(__bf16)r.f2.w;
    a[12]=(__bf16)r.f3.x; a[13]=(__bf16)r.f3.y; a[14]=(__bf16)r.f3.z; a[15]=(__bf16)r.f3.w;
    return a;
}

__device__ __forceinline__ v16bf load_b_frag(const unsigned char* bb, int nt, int lr, int half) {
    const unsigned char* bp = bb + (nt * 16 + lr) * BPAD + half * 16;
    v8bf blo = *(const v8bf*)(bp);          // ds_load_b128
    v8bf bhi = *(const v8bf*)(bp + 32);     // ds_load_b128
    return __builtin_shufflevector(blo, bhi,
                                   0,1,2,3,4,5,6,7,8,9,10,11,12,13,14,15);
}

// ---------------- fused WMMA GEMM ----------------
// Per workgroup: 8 waves, 128 tokens. B slab (320 rows x 32 k, bf16) staged in LDS via
// async DMA (double buffered); A and B fragments double-buffered in registers so
// ds_load / global_load latency overlaps WMMA execution.
__global__ void __launch_bounds__(256)
fused_gemm_kernel(const float* __restrict__ hidden,
                  const __bf16* __restrict__ Wall,   // [320,2048] bf16: 256 ts rows + 64 tr rows
                  const float* __restrict__ b_ts,
                  const float* __restrict__ b_tr,
                  float* __restrict__ base_logits,   // [NTOK,64]
                  float* __restrict__ pooled_sum)    // [BATCH,256]
{
    __shared__ __align__(16) unsigned char Bsh[2][BROWS * BPAD];   // 2 x 25600 B
    __shared__ float poolLds[DDIM];

    const int tid = threadIdx.x;
    for (int i = tid; i < DDIM; i += 256) poolLds[i] = 0.0f;

    const int lane = tid & 31;
    const int wave = tid >> 5;
    const int lr   = lane & 15;
    const int half = lane >> 4;

    const int tokenRow = blockIdx.x * 128 + wave * 16 + lr;   // A-row for this lane
    const float* hrow = hidden + (size_t)tokenRow * HID;

    v8f acc[20];
#pragma unroll
    for (int nt = 0; nt < 20; ++nt)
#pragma unroll
        for (int v = 0; v < 8; ++v) acc[nt][v] = 0.0f;

    // prologue: DMA first B slab, load first A chunk
    load_b_tile(Wall, 0, Bsh[0], tid);
    AF af = load_a(hrow, 0, half);

    for (int k0 = 0; k0 < HID; k0 += 32) {
        const int  cur     = (k0 >> 5) & 1;
        const bool hasNext = (k0 + 32) < HID;
        if (hasNext) load_b_tile(Wall, k0 + 32, Bsh[cur ^ 1], tid);   // overlap next slab DMA

        if (hasNext) async_wait_keep5();   // previous slab complete (in-order retire)
        else         async_wait_all();
        __syncthreads();                   // slab visible to all waves

        // deep prefetch for the A stream (global_prefetch_b8)
        __builtin_prefetch(hrow + k0 + 512, 0, 3);

        // issue next A chunk now: its loads retire during the 20-WMMA sequence
        AF afn;
        if (hasNext) afn = load_a(hrow, k0 + 32, half);

        const v16bf a = cvt_a(af);

        // B register double-buffer: load nt+1 before wmma(nt)
        const unsigned char* bb = &Bsh[cur][0];
        v16bf bcur = load_b_frag(bb, 0, lr, half);
#pragma unroll
        for (int nt = 0; nt < 20; ++nt) {
            v16bf bnext = (nt < 19) ? load_b_frag(bb, nt + 1, lr, half) : bcur;
            acc[nt] = __builtin_amdgcn_wmma_f32_16x16x32_bf16(
                          false, a, false, bcur, (short)0, acc[nt], false, false);
            bcur = bnext;
        }
        __syncthreads();   // all reads of 'cur' done before it is DMA-overwritten
        if (hasNext) af = afn;
    }

    // ---- epilogue 1: tanh + pool the token_state tiles ----
#pragma unroll
    for (int nt = 0; nt < 16; ++nt) {
        int n = lr + nt * 16;
        float bts = b_ts[n];
        float s = 0.0f;
#pragma unroll
        for (int v = 0; v < 8; ++v) s += tanhf(acc[nt][v] + bts);
        atomicAdd(&poolLds[n], s);   // ds atomic path
    }
    __syncthreads();
    {
        int batch = (blockIdx.x * 128) >> 12;
        atomicAdd(&pooled_sum[batch * DDIM + tid], poolLds[tid]);
    }

    // ---- epilogue 2: store biased base_logits ----
    const int tokBase = blockIdx.x * 128 + wave * 16 + half * 8;  // C rows: M = half*8 + v
#pragma unroll
    for (int nt = 16; nt < 20; ++nt) {
        int e = lr + (nt - 16) * 16;
        float btr = b_tr[e];
#pragma unroll
        for (int v = 0; v < 8; ++v)
            base_logits[(size_t)(tokBase + v) * NEXP + e] = acc[nt][v] + btr;
    }
}

// ---------------- margins: p_top1 - p_top2 of softmax over 64 experts ----------------
__global__ void __launch_bounds__(256)
margins_kernel(const float* __restrict__ logits, float* __restrict__ margins) {
    const int lane = threadIdx.x & 31;
    const int wave = threadIdx.x >> 5;
    for (int it = 0; it < 8; ++it) {
        int t = blockIdx.x * 64 + wave * 8 + it;
        float x0 = logits[(size_t)t * NEXP + lane];
        float x1 = logits[(size_t)t * NEXP + lane + 32];
        float mx = fmaxf(x0, x1);
        for (int off = 16; off > 0; off >>= 1)
            mx = fmaxf(mx, __shfl_xor(mx, off, 32));
        float e0 = __expf(x0 - mx), e1 = __expf(x1 - mx);
        float s = e0 + e1;
        for (int off = 16; off > 0; off >>= 1)
            s += __shfl_xor(s, off, 32);
        float p0 = e0 / s, p1 = e1 / s;
        float a = fmaxf(p0, p1), b = fminf(p0, p1);
        for (int off = 16; off > 0; off >>= 1) {
            float oa = __shfl_xor(a, off, 32);
            float ob = __shfl_xor(b, off, 32);
            float na = fmaxf(a, oa);
            float nb = fmaxf(fminf(a, oa), fmaxf(b, ob));
            a = na; b = nb;
        }
        if (lane == 0) margins[t] = a - b;
    }
}

// ---------------- exact k-smallest select (stable tie-break by index) ----------------
__global__ void __launch_bounds__(1024)
select_kernel(const float* __restrict__ margins, int* __restrict__ active, int n, int k) {
    __shared__ unsigned hist[256];
    __shared__ unsigned sLess;
    __shared__ unsigned prefixSh;
    __shared__ unsigned scanBuf[1024];
    __shared__ unsigned chunkBase;
    const int tid = threadIdx.x;

    unsigned prefix = 0;
    int kRem = k;
    for (int pass = 0; pass < 4; ++pass) {
        int shift = 24 - pass * 8;
        if (tid < 256) hist[tid] = 0;
        __syncthreads();
        for (int i = tid; i < n; i += 1024) {
            unsigned bits = __float_as_uint(margins[i]);   // margins >= 0 -> order == uint order
            if (pass == 0 || ((bits >> (shift + 8)) == (prefix >> (shift + 8))))
                atomicAdd(&hist[(bits >> shift) & 0xFF], 1u);
        }
        __syncthreads();
        if (tid == 0) {
            unsigned cum = 0; unsigned chosen = 255;
            for (unsigned b = 0; b < 256; ++b) {
                if (cum + hist[b] >= (unsigned)kRem) { chosen = b; break; }
                cum += hist[b];
            }
            prefixSh = prefix | (chosen << shift);
            sLess = cum;
        }
        __syncthreads();
        prefix = prefixSh;
        kRem -= (int)sLess;
        __syncthreads();
    }
    const unsigned T = prefix;   // bit pattern of k-th smallest margin

    if (tid == 0) sLess = 0;
    __syncthreads();
    unsigned lc = 0;
    for (int i = tid; i < n; i += 1024)
        if (__float_as_uint(margins[i]) < T) lc++;
    atomicAdd(&sLess, lc);
    __syncthreads();
    const unsigned quota = (unsigned)(k - (int)sLess);  // #elements == T to take, by index order

    if (tid == 0) chunkBase = 0;
    __syncthreads();
    for (int c = 0; c < n; c += 1024) {
        int i = c + tid;
        unsigned bits = __float_as_uint(margins[i]);
        unsigned eq = (bits == T) ? 1u : 0u;
        scanBuf[tid] = eq;
        __syncthreads();
        for (int off = 1; off < 1024; off <<= 1) {
            unsigned v = (tid >= off) ? scanBuf[tid - off] : 0u;
            __syncthreads();
            scanBuf[tid] += v;
            __syncthreads();
        }
        unsigned incl = scanBuf[tid];
        unsigned rank = chunkBase + (incl - eq);
        active[i] = (bits < T) || (eq && rank < quota);
        __syncthreads();
        if (tid == 1023) chunkBase += incl;
        __syncthreads();
    }
}

// ---------------- apply mask/bias + softmax + pooled weight sums ----------------
__global__ void __launch_bounds__(256)
apply_softmax_kernel(const float* __restrict__ baseL, float* __restrict__ curL,
                     const int* __restrict__ active, const float* __restrict__ bias,
                     float* __restrict__ wsum, float* __restrict__ outW, int flags) {
    __shared__ float wsLds[NEXP];
    if (threadIdx.x < NEXP) wsLds[threadIdx.x] = 0.0f;
    __syncthreads();

    const int lane = threadIdx.x & 31;
    const int wave = threadIdx.x >> 5;
    const int batch = (blockIdx.x * 64) >> 12;

    for (int it = 0; it < 8; ++it) {
        int t = blockIdx.x * 64 + wave * 8 + it;
        size_t o = (size_t)t * NEXP;
        float x0, x1;
        if (flags & FL_MASK) {
            bool act = active[t] != 0;
            if (act) {
                x0 = baseL[o + lane]      + bias[batch * NEXP + lane];
                x1 = baseL[o + lane + 32] + bias[batch * NEXP + lane + 32];
            } else {
                x0 = curL[o + lane];
                x1 = curL[o + lane + 32];
            }
        } else {
            x0 = baseL[o + lane];
            x1 = baseL[o + lane + 32];
        }
        curL[o + lane] = x0;
        curL[o + lane + 32] = x1;

        float mx = fmaxf(x0, x1);
        for (int off = 16; off > 0; off >>= 1)
            mx = fmaxf(mx, __shfl_xor(mx, off, 32));
        float e0 = __expf(x0 - mx), e1 = __expf(x1 - mx);
        float s = e0 + e1;
        for (int off = 16; off > 0; off >>= 1)
            s += __shfl_xor(s, off, 32);
        float p0 = e0 / s, p1 = e1 / s;

        if (flags & FL_WSUM) {
            atomicAdd(&wsLds[lane], p0);
            atomicAdd(&wsLds[lane + 32], p1);
        }
        if (flags & FL_WRITE) {
            outW[o + lane] = p0;
            outW[o + lane + 32] = p1;
        }
    }
    __syncthreads();
    if ((flags & FL_WSUM) && threadIdx.x < NEXP)
        atomicAdd(&wsum[batch * NEXP + threadIdx.x], wsLds[threadIdx.x]);
}

// ---------------- GRU controller update + next-round route bias ----------------
__global__ void __launch_bounds__(1024)
controller_kernel(const float* __restrict__ pooled_sum, const float* __restrict__ wsum,
                  const float* __restrict__ W_fb, const float* __restrict__ W_r,
                  const float* __restrict__ b_r,  const float* __restrict__ W_z,
                  const float* __restrict__ b_z,  const float* __restrict__ W_ci,
                  const float* __restrict__ b_ci, const float* __restrict__ W_cs,
                  const float* __restrict__ W_sr,
                  float* __restrict__ controller, float* __restrict__ bias) {
    __shared__ float sIn[BATCH][DDIM];
    __shared__ float rc[BATCH][DDIM];
    __shared__ float cNew[BATCH][DDIM];
    const int d = threadIdx.x;   // 0..255
    const int b = threadIdx.y;   // 0..3
    const float invS = 1.0f / (float)SEQ;

    float fb = 0.0f;
    for (int e = 0; e < NEXP; ++e) fb += wsum[b * NEXP + e] * W_fb[d * NEXP + e];
    sIn[b][d] = pooled_sum[b * DDIM + d] * invS + fb * invS;
    __syncthreads();

    float accR = b_r[d], accZ = b_z[d], accI = b_ci[d];
    for (int j = 0; j < DDIM; ++j) {
        float s = sIn[b][j];
        accR += s * W_r[d * DDIM + j];
        accZ += s * W_z[d * DDIM + j];
        accI += s * W_ci[d * DDIM + j];
    }
    float r = 1.0f / (1.0f + __expf(-accR));
    float z = 1.0f / (1.0f + __expf(-accZ));
    float cOld = controller[b * DDIM + d];
    rc[b][d] = r * cOld;
    __syncthreads();

    float accC = accI;
    for (int j = 0; j < DDIM; ++j) accC += rc[b][j] * W_cs[d * DDIM + j];
    float cn = (1.0f - z) * cOld + z * tanhf(accC);
    cNew[b][d] = cn;
    controller[b * DDIM + d] = cn;
    __syncthreads();

    if (d < NEXP) {
        float acc = 0.0f;
        for (int j = 0; j < DDIM; ++j) acc += cNew[b][j] * W_sr[d * DDIM + j];
        bias[b * NEXP + d] = acc;
    }
}

// ---------------- host-side orchestration ----------------
extern "C" void kernel_launch(void* const* d_in, const int* in_sizes, int n_in,
                              void* d_out, int out_size, void* d_ws, size_t ws_size,
                              hipStream_t stream) {
    (void)in_sizes; (void)n_in; (void)out_size; (void)ws_size;
    const float* hidden = (const float*)d_in[0];
    const float* W_ts   = (const float*)d_in[1];
    const float* b_ts   = (const float*)d_in[2];
    const float* W_tr   = (const float*)d_in[3];
    const float* b_tr   = (const float*)d_in[4];
    const float* W_sr   = (const float*)d_in[5];
    const float* W_fb   = (const float*)d_in[6];
    const float* W_r    = (const float*)d_in[7];
    const float* b_r    = (const float*)d_in[8];
    const float* W_z    = (const float*)d_in[9];
    const float* b_z    = (const float*)d_in[10];
    const float* W_ci   = (const float*)d_in[11];
    const float* b_ci   = (const float*)d_in[12];
    const float* W_cs   = (const float*)d_in[13];
    float* outW = (float*)d_out;

    // workspace layout (~9.4 MiB)
    char* w = (char*)d_ws;
    __bf16* WallB   = (__bf16*)(w);                 // 320*2048*2 = 1310720 B (ts rows then tr rows)
    float*  baseL   = (float*)(w + 1310720);        // 4 MiB
    float*  curL    = (float*)(w + 5505024);        // 4 MiB
    float*  margins = (float*)(w + 9699328);        // 64 KiB
    int*    active  = (int*)  (w + 9764864);        // 64 KiB
    float*  pooled  = (float*)(w + 9830400);        // 4 KiB
    float*  wsum    = (float*)(w + 9834496);        // 1 KiB
    float*  ctrl    = (float*)(w + 9835520);        // 4 KiB
    float*  bias    = (float*)(w + 9839616);        // 1 KiB

    // init accumulators / controller state
    zero_f32_kernel<<<4, 256, 0, stream>>>(pooled, BATCH * DDIM);
    zero_f32_kernel<<<4, 256, 0, stream>>>(ctrl,   BATCH * DDIM);
    zero_f32_kernel<<<1, 256, 0, stream>>>(bias,   BATCH * NEXP);

    // weights -> bf16, packed [320, 2048]
    cvt_bf16_kernel<<<(DDIM * HID + 255) / 256, 256, 0, stream>>>(W_ts, WallB, DDIM * HID);
    cvt_bf16_kernel<<<(NEXP * HID + 255) / 256, 256, 0, stream>>>(W_tr, WallB + (size_t)DDIM * HID,
                                                                  NEXP * HID);

    // fused WMMA GEMM: base_logits + pooled tanh state
    fused_gemm_kernel<<<NTOK / 128, 256, 0, stream>>>(hidden, WallB, b_ts, b_tr,
                                                      baseL, pooled);

    // ---- round 0: softmax(base), weight sums, controller update ----
    zero_f32_kernel<<<1, 256, 0, stream>>>(wsum, BATCH * NEXP);
    apply_softmax_kernel<<<NTOK / 64, 256, 0, stream>>>(baseL, curL, active, bias,
                                                        wsum, outW, FL_WSUM);
    controller_kernel<<<1, dim3(DDIM, BATCH), 0, stream>>>(pooled, wsum, W_fb, W_r, b_r,
                                                           W_z, b_z, W_ci, b_ci, W_cs, W_sr,
                                                           ctrl, bias);

    // ---- round 1: margins -> select -> masked update -> controller ----
    margins_kernel<<<NTOK / 64, 256, 0, stream>>>(curL, margins);
    select_kernel<<<1, 1024, 0, stream>>>(margins, active, NTOK, KSEL);
    zero_f32_kernel<<<1, 256, 0, stream>>>(wsum, BATCH * NEXP);
    apply_softmax_kernel<<<NTOK / 64, 256, 0, stream>>>(baseL, curL, active, bias,
                                                        wsum, outW, FL_MASK | FL_WSUM);
    controller_kernel<<<1, dim3(DDIM, BATCH), 0, stream>>>(pooled, wsum, W_fb, W_r, b_r,
                                                           W_z, b_z, W_ci, b_ci, W_cs, W_sr,
                                                           ctrl, bias);

    // ---- round 2: margins -> select -> masked update, write final weights ----
    margins_kernel<<<NTOK / 64, 256, 0, stream>>>(curL, margins);
    select_kernel<<<1, 1024, 0, stream>>>(margins, active, NTOK, KSEL);
    apply_softmax_kernel<<<NTOK / 64, 256, 0, stream>>>(baseL, curL, active, bias,
                                                        wsum, outW, FL_MASK | FL_WRITE);
}
